// GDM_71605694759232
// MI455X (gfx1250) — matrix operand order repaired and checked by reading, not compiled, not run on hardware
//
#include <hip/hip_runtime.h>
#include <hip/hip_bf16.h>

typedef __bf16 bf16_t;
typedef __attribute__((ext_vector_type(8)))  __bf16 bf16x8;
typedef __attribute__((ext_vector_type(16))) __bf16 bf16x16;
typedef __attribute__((ext_vector_type(8)))  float  f32x8;

constexpr int V   = 32000;
constexpr int D   = 512;
constexpr int H   = 8;
constexpr int NL  = 2;
constexpr int FF  = 2048;
constexpr int BB  = 4;
constexpr int S   = 512;
constexpr int SP1 = 513;   // S+1
constexpr int SP  = 528;   // padded to 33 tiles of 16
constexpr float LN_EPS = 1e-5f;

// ---------------------------------------------------------------- fragments
// A-matrix (16x32 bf16, MxK): lane m=lane&15; VGPR0..3 K=h*8.., VGPR4..7 K=16+h*8..
__device__ __forceinline__ bf16x16 frag_a(const bf16_t* base, int row, int ld, int kb, int hf) {
  const bf16_t* p = base + (size_t)row * ld + kb + hf * 8;
  bf16x8 lo = *(const bf16x8*)p;
  bf16x8 hi = *(const bf16x8*)(p + 16);
  return __builtin_shufflevector(lo, hi, 0,1,2,3,4,5,6,7,8,9,10,11,12,13,14,15);
}
// B-matrix (32x16 bf16, KxN) loaded from Bt (N rows, K contiguous):
// lane n=lane&15; VGPR0..3 K=h*8.., VGPR4..7 K=16+h*8.. (same as A)
__device__ __forceinline__ bf16x16 frag_b(const bf16_t* base, int row, int ld, int kb, int hf) {
  const bf16_t* p = base + (size_t)row * ld + kb + hf * 8;
  bf16x8 lo = *(const bf16x8*)p;
  bf16x8 hi = *(const bf16x8*)(p + 16);
  return __builtin_shufflevector(lo, hi, 0,1,2,3,4,5,6,7,8,9,10,11,12,13,14,15);
}
__device__ __forceinline__ f32x8 wmma_bf(bf16x16 a, bf16x16 b, f32x8 c) {
  return __builtin_amdgcn_wmma_f32_16x16x32_bf16(false, a, false, b, (short)0, c, false, false);
}

// ---------------------------------------------------------------- utilities
__global__ void tk_zero(float* p, int n) {
  for (int i = blockIdx.x * blockDim.x + threadIdx.x; i < n; i += gridDim.x * blockDim.x) p[i] = 0.f;
}
__global__ void tk_cvt_bf16(const float* __restrict__ s, bf16_t* __restrict__ d, int n) {
  for (int i = blockIdx.x * blockDim.x + threadIdx.x; i < n; i += gridDim.x * blockDim.x)
    d[i] = (bf16_t)s[i];
}
// transpose f32 (rows x cols) -> bf16 (cols x rows); 32x32 tiles, 256 threads
__global__ void tk_transpose_bf16(const float* __restrict__ X, bf16_t* __restrict__ XT,
                                  int rows, int cols) {
  __shared__ float t[32][33];
  int r0 = blockIdx.x * 32, c0 = blockIdx.y * 32;
  int tid = threadIdx.x, cc = tid & 31, rr4 = (tid >> 5) * 4;
  #pragma unroll
  for (int j = 0; j < 4; ++j) t[rr4 + j][cc] = X[(size_t)(r0 + rr4 + j) * cols + c0 + cc];
  __syncthreads();
  #pragma unroll
  for (int j = 0; j < 4; ++j) XT[(size_t)(c0 + rr4 + j) * rows + r0 + cc] = (bf16_t)t[cc][rr4 + j];
}
// e[b,s,:] = W_e[x[b,s],:]   (block = 128 threads, one row)
__global__ void tk_embed(const int* __restrict__ x, const float* __restrict__ We,
                         float* __restrict__ e) {
  int rid = blockIdx.x;
  int tok = x[rid];
  const float4* src = (const float4*)(We + (size_t)tok * D);
  float4* dst = (float4*)(e + (size_t)rid * D);
  dst[threadIdx.x] = src[threadIdx.x];
}
// out[b,d] += sum_r X[b, r0..r0+63, d]  (atomic partials; 256 thr handle 512 cols)
__global__ void tk_colsum(const float* __restrict__ X, float* __restrict__ out, int rowsTotal) {
  int b = blockIdx.y;
  const float* Xb = X + (size_t)b * rowsTotal * D;
  float* ob = out + (size_t)b * D;
  int r0 = blockIdx.x * 64, tid = threadIdx.x;
  float a0 = 0.f, a1 = 0.f;
  for (int r = 0; r < 64; ++r) {
    a0 += Xb[(size_t)(r0 + r) * D + tid];
    a1 += Xb[(size_t)(r0 + r) * D + tid + 256];
  }
  atomicAdd(ob + tid, a0);
  atomicAdd(ob + tid + 256, a1);
}
// Q[h,q,:] = W_p[q]*W_q_diag[h] (q<513, else 0);  K[h,k,:] = W_p[k]*W_k_diag[h] (k<512)
__global__ void tk_qk_prep(const float* __restrict__ Wp, const float* __restrict__ Wq,
                           const float* __restrict__ Wk, bf16_t* __restrict__ Qb,
                           bf16_t* __restrict__ Kb) {
  int q = blockIdx.x, h = blockIdx.y, tid = threadIdx.x;
  for (int d = tid; d < D; d += 256) {
    float p = (q < SP1) ? Wp[(size_t)q * D + d] : 0.f;
    Qb[((size_t)h * SP + q) * D + d] = (bf16_t)((q < SP1) ? p * Wq[h * D + d] : 0.f);
    if (q < S) Kb[((size_t)h * S + q) * D + d] = (bf16_t)(p * Wk[h * D + d]);
  }
}
// softmax over k of scores row (h,q); pad rows q>=513 -> 0
__global__ void tk_softmax(const float* __restrict__ scores, bf16_t* __restrict__ attn) {
  int row = blockIdx.x, q = row % SP, tid = threadIdx.x;
  bf16_t* out = attn + (size_t)row * S;
  if (q >= SP1) { out[tid] = (bf16_t)0.f; out[tid + 256] = (bf16_t)0.f; return; }
  const float* x = scores + (size_t)row * S;
  __shared__ float red[256];
  float a = x[tid], b = x[tid + 256];
  red[tid] = fmaxf(a, b); __syncthreads();
  for (int o = 128; o > 0; o >>= 1) { if (tid < o) red[tid] = fmaxf(red[tid], red[tid + o]); __syncthreads(); }
  float mx = red[0]; __syncthreads();
  float ea = expf(a - mx), eb = expf(b - mx);
  red[tid] = ea + eb; __syncthreads();
  for (int o = 128; o > 0; o >>= 1) { if (tid < o) red[tid] += red[tid + o]; __syncthreads(); }
  float inv = 1.f / red[0];
  out[tid] = (bf16_t)(ea * inv);
  out[tid + 256] = (bf16_t)(eb * inv);
}
// diffT[b][d][s] = e[b,s,d] - E(b,s,d);  E = Ewe or colsum(W_e)*invV
__global__ void tk_diff_transpose(const float* __restrict__ e, const float* __restrict__ Ewe,
                                  const float* __restrict__ WeCol, float invV, int useEwe,
                                  bf16_t* __restrict__ dT) {
  __shared__ float t[32][33];
  int b = blockIdx.z, s0 = blockIdx.y * 32, d0 = blockIdx.x * 32;
  int tid = threadIdx.x, cc = tid & 31, rr4 = (tid >> 5) * 4;
  #pragma unroll
  for (int j = 0; j < 4; ++j) {
    int sl = rr4 + j;
    size_t idx = ((size_t)b * S + s0 + sl) * D + d0 + cc;
    float Ev = useEwe ? Ewe[idx] : WeCol[d0 + cc] * invV;
    t[sl][cc] = e[idx] - Ev;
  }
  __syncthreads();
  #pragma unroll
  for (int j = 0; j < 4; ++j) {
    int dl = rr4 + j;
    dT[((size_t)b * D + d0 + dl) * S + s0 + cc] = (bf16_t)t[cc][dl];
  }
}
// diffsum[b,d] = sum_s diffT[b][d][s]   (one wave per row)
__global__ void tk_diffsum(const bf16_t* __restrict__ dT, float* __restrict__ ds) {
  int w = threadIdx.x >> 5, lane = threadIdx.x & 31;
  int row = blockIdx.x * 8 + w;
  const bf16_t* p = dT + (size_t)row * S + lane * 16;
  float s = 0.f;
  #pragma unroll
  for (int j = 0; j < 16; ++j) s += (float)p[j];
  #pragma unroll
  for (int o = 16; o > 0; o >>= 1) s += __shfl_xor(s, o, 32);
  if (lane == 0) ds[row] = s;
}

// ---------------------------------------------------------------- generic GEMM
// C[M,N] = A[M,K] @ Bt[N,K]^T  bf16 in, f32 acc. Per wave: 4x4 tiles of 16x16.
// Block: 4 waves (2m x 2n) -> 128x128 tile. K-loop is double-buffered so WMMAs
// on one fragment set overlap the global loads of the next.
// EPI: 0=f32, 1=gelu->bf16, 2=f32+bf16
template <int EPI>
__global__ __launch_bounds__(128) void tk_gemm(
    const bf16_t* __restrict__ A, const bf16_t* __restrict__ Bt,
    float* __restrict__ C, bf16_t* __restrict__ Cb,
    int M, int N, int K, int lda, int ldb, int ldc,
    long sA, long sB, long sC) {
  int z = blockIdx.z;
  const bf16_t* Ab = A + (size_t)z * sA;
  const bf16_t* Bb = Bt + (size_t)z * sB;
  int w = threadIdx.x >> 5, lane = threadIdx.x & 31, ln = lane & 15, hf = lane >> 4;
  int mt0 = blockIdx.y * 8 + (w >> 1) * 4;
  int nt0 = blockIdx.x * 8 + (w & 1) * 4;
  f32x8 acc[4][4] = {};
  int arow[4], brow[4];
  #pragma unroll
  for (int mt = 0; mt < 4; ++mt)
    arow[mt] = (((mt0 + mt) * 16 < M) ? (mt0 + mt) * 16 : 0) + ln;  // clamp pads to row 0
  #pragma unroll
  for (int nt = 0; nt < 4; ++nt) brow[nt] = (nt0 + nt) * 16 + ln;

  auto load_set = [&](int kb, bf16x16* af, bf16x16* bf) {
    #pragma unroll
    for (int mt = 0; mt < 4; ++mt) af[mt] = frag_a(Ab, arow[mt], lda, kb, hf);
    #pragma unroll
    for (int nt = 0; nt < 4; ++nt) bf[nt] = frag_b(Bb, brow[nt], ldb, kb, hf);
  };
  auto mma_set = [&](bf16x16* af, bf16x16* bf) {
    #pragma unroll
    for (int nt = 0; nt < 4; ++nt)
      #pragma unroll
      for (int mt = 0; mt < 4; ++mt)
        acc[mt][nt] = wmma_bf(af[mt], bf[nt], acc[mt][nt]);
  };

  bf16x16 afX[4], bfX[4], afY[4], bfY[4];
  load_set(0, afX, bfX);
  for (int kb = 0; kb < K; kb += 64) {       // K is a multiple of 64
    load_set(kb + 32, afY, bfY);
    mma_set(afX, bfX);
    if (kb + 64 < K) load_set(kb + 64, afX, bfX);
    mma_set(afY, bfY);
  }

  #pragma unroll
  for (int mt = 0; mt < 4; ++mt) {
    if ((mt0 + mt) * 16 >= M) continue;   // wave-uniform store guard
    #pragma unroll
    for (int nt = 0; nt < 4; ++nt) {
      int ncol = (nt0 + nt) * 16 + ln;
      #pragma unroll
      for (int r = 0; r < 8; ++r) {
        int m = (mt0 + mt) * 16 + r + 8 * hf;
        size_t idx = (size_t)z * sC + (size_t)m * ldc + ncol;
        float v = acc[mt][nt][r];
        if constexpr (EPI == 0) {
          C[idx] = v;
        } else if constexpr (EPI == 1) {
          Cb[idx] = (bf16_t)(0.5f * v * (1.f + erff(v * 0.70710678118654752f)));
        } else {
          C[idx] = v;
          Cb[idx] = (bf16_t)v;
        }
      }
    }
  }
}

// ---------------------------------------------------------------- delta update
// f_k[b,q,e] += ( sum_h A_lr[h]*Wv[h,e]*(attn[h]@diff[b])[q,e] + B_lr*diffsum[b,e] ) / S
// h inner with 8 persistent accumulators; all 8 A-frags issued before the WMMA set.
__global__ __launch_bounds__(32) void tk_delta_update(
    const bf16_t* __restrict__ attn, const bf16_t* __restrict__ dT,
    const float* __restrict__ Wv, const float* __restrict__ Alr,
    const float* __restrict__ dsum, const float* __restrict__ Blr,
    float* __restrict__ fk) {
  int et = blockIdx.x, qt = blockIdx.y, b = blockIdx.z;
  int lane = threadIdx.x & 31, ln = lane & 15, hf = lane >> 4;
  int ncol = et * 16 + ln;
  const bf16_t* dTb = dT + (size_t)b * D * S;   // rows = e, ld = S
  int arow = qt * 16 + ln, brow = et * 16 + ln;
  f32x8 acc[H] = {};
  #pragma unroll 2
  for (int kb = 0; kb < S; kb += 32) {
    bf16x16 bfr = frag_b(dTb, brow, S, kb, hf);
    bf16x16 af[H];
    #pragma unroll
    for (int h = 0; h < H; ++h) af[h] = frag_a(attn + (size_t)h * SP * S, arow, S, kb, hf);
    #pragma unroll
    for (int h = 0; h < H; ++h) acc[h] = wmma_bf(af[h], bfr, acc[h]);
  }
  float tot[8] = {};
  #pragma unroll
  for (int h = 0; h < H; ++h) {
    float sc = Alr[h] * Wv[h * D + ncol];
    #pragma unroll
    for (int r = 0; r < 8; ++r) tot[r] += sc * acc[h][r];
  }
  float bterm = Blr[0] * dsum[b * D + ncol];
  const float invS = 1.f / (float)S;
  #pragma unroll
  for (int r = 0; r < 8; ++r) {
    int q = qt * 16 + r + 8 * hf;
    if (q < SP1) {
      size_t idx = ((size_t)b * SP + q) * D + ncol;
      fk[idx] += (tot[r] + bterm) * invS;
    }
  }
}

// ---------------------------------------------------------------- layernorm
__global__ void tk_layernorm(const float* __restrict__ fk, const float* __restrict__ g,
                             bf16_t* __restrict__ out) {
  __shared__ float red[256];
  int row = blockIdx.x, tid = threadIdx.x;
  const float* x = fk + (size_t)row * D;
  float a = x[tid], b = x[tid + 256];
  red[tid] = a + b; __syncthreads();
  for (int o = 128; o > 0; o >>= 1) { if (tid < o) red[tid] += red[tid + o]; __syncthreads(); }
  float mu = red[0] / (float)D; __syncthreads();
  float d0 = a - mu, d1 = b - mu;
  red[tid] = d0 * d0 + d1 * d1; __syncthreads();
  for (int o = 128; o > 0; o >>= 1) { if (tid < o) red[tid] += red[tid + o]; __syncthreads(); }
  float rs = rsqrtf(red[0] / (float)D + LN_EPS);
  out[(size_t)row * D + tid] = (bf16_t)(d0 * rs * g[tid]);
  out[(size_t)row * D + tid + 256] = (bf16_t)(d1 * rs * g[tid + 256]);
}

// ---------------------------------------------------------------- fused E_W_e (layer 1)
// Ewe[b,s,:] = sum_v exp(clip(f.We_v)) * We_v / (sum_v exp + 1e-8); flash style, 256 thr.
// 16x512 query tile lives in registers (16 A-fragments); B loads batched in sets of 4.
__global__ __launch_bounds__(256) void tk_flash_ewe(
    const bf16_t* __restrict__ fkb, const bf16_t* __restrict__ We,
    const bf16_t* __restrict__ WeT, float* __restrict__ Ewe) {
  __shared__ __align__(16) bf16_t Tlds[16][128];
  __shared__ float red[16][16];
  __shared__ float den16[16];
  int bidx = blockIdx.x;               // 0..127
  int b = bidx >> 5, st = bidx & 31;
  int tid = threadIdx.x;
  int w = tid >> 5, lane = tid & 31, ln = lane & 15, hf = lane >> 4;
  bf16x16 afr[16];
  int qrow = b * SP + st * 16 + ln;
  #pragma unroll
  for (int kk = 0; kk < 16; ++kk) afr[kk] = frag_a(fkb, qrow, D, kk * 32, hf);
  float denacc = 0.f;
  f32x8 acc2[4] = {};
  for (int c = 0; c < V / 128; ++c) {
    { // stage 1: Z = F @ We^T for this 128-v chunk; wave w owns 16 columns
      f32x8 z = {};
      int vrow = c * 128 + w * 16 + ln;
      #pragma unroll
      for (int kk = 0; kk < 16; kk += 4) {
        bf16x16 bq[4];
        #pragma unroll
        for (int j = 0; j < 4; ++j) bq[j] = frag_b(We, vrow, D, (kk + j) * 32, hf);
        #pragma unroll
        for (int j = 0; j < 4; ++j) z = wmma_bf(afr[kk + j], bq[j], z);
      }
      #pragma unroll
      for (int r = 0; r < 8; ++r) {
        float t = expf(fminf(fmaxf(z[r], -10.f), 10.f));
        Tlds[r + 8 * hf][w * 16 + ln] = (bf16_t)t;
      }
    }
    __syncthreads();
    { // denominator partials
      int m = tid >> 4, part = tid & 15;
      float s = 0.f;
      #pragma unroll
      for (int j = 0; j < 8; ++j) s += (float)Tlds[m][part * 8 + j];
      denacc += s;
    }
    // stage 2: acc += T @ We ; wave w owns 64 d-columns
    #pragma unroll
    for (int kk = 0; kk < 4; ++kk) {
      bf16x16 ta = frag_a(&Tlds[0][0], ln, 128, kk * 32, hf);
      bf16x16 bw[4];
      #pragma unroll
      for (int ntl = 0; ntl < 4; ++ntl)
        bw[ntl] = frag_b(WeT, w * 64 + ntl * 16 + ln, V, c * 128 + kk * 32, hf);
      #pragma unroll
      for (int ntl = 0; ntl < 4; ++ntl) acc2[ntl] = wmma_bf(ta, bw[ntl], acc2[ntl]);
    }
    __syncthreads();
  }
  red[tid >> 4][tid & 15] = denacc;
  __syncthreads();
  if (tid < 16) {
    float s = 0.f;
    for (int j = 0; j < 16; ++j) s += red[tid][j];
    den16[tid] = s + 1e-8f;
  }
  __syncthreads();
  #pragma unroll
  for (int ntl = 0; ntl < 4; ++ntl) {
    int d = w * 64 + ntl * 16 + ln;
    #pragma unroll
    for (int r = 0; r < 8; ++r) {
      int m = r + 8 * hf;
      Ewe[((size_t)b * S + st * 16 + m) * D + d] = acc2[ntl][r] / den16[m];
    }
  }
}

// ---------------------------------------------------------------- launch
extern "C" void kernel_launch(void* const* d_in, const int* in_sizes, int n_in,
                              void* d_out, int out_size, void* d_ws, size_t ws_size,
                              hipStream_t stream) {
  const int*   x    = (const int*)d_in[0];
  const float* W_e  = (const float*)d_in[1];
  const float* W_p  = (const float*)d_in[2];
  const float* W_q  = (const float*)d_in[3];
  const float* W_k  = (const float*)d_in[4];
  const float* W_v  = (const float*)d_in[5];
  const float* A_lr = (const float*)d_in[6];
  const float* B_lr = (const float*)d_in[7];
  const float* lng  = (const float*)d_in[8];
  const float* ff1  = (const float*)d_in[9];
  const float* ff2  = (const float*)d_in[10];
  float* out = (float*)d_out;

  char* ws = (char*)d_ws;
  size_t off = 0;
  auto alloc = [&](size_t bytes) -> char* {
    char* p = ws + off;
    off += (bytes + 255) & ~(size_t)255;
    return p;
  };
  bf16_t* We_b  = (bf16_t*)alloc((size_t)V * D * 2);
  bf16_t* WeT_b = (bf16_t*)alloc((size_t)V * D * 2);
  bf16_t* w1_b  = (bf16_t*)alloc((size_t)NL * FF * D * 2);
  bf16_t* w2_b  = (bf16_t*)alloc((size_t)NL * D * FF * 2);
  float*  e_f   = (float*)alloc((size_t)BB * S * D * 4);
  float*  esum  = (float*)alloc((size_t)BB * D * 4);
  float*  WeCol = (float*)alloc((size_t)D * 4);
  bf16_t* Qb    = (bf16_t*)alloc((size_t)H * SP * D * 2);
  bf16_t* Kb    = (bf16_t*)alloc((size_t)H * S * D * 2);
  float*  scr   = (float*)alloc((size_t)H * SP * S * 4);
  bf16_t* attn  = (bf16_t*)alloc((size_t)H * SP * S * 2);
  float*  fk    = (float*)alloc((size_t)BB * SP * D * 4);
  bf16_t* fk_b  = (bf16_t*)alloc((size_t)BB * SP * D * 2);
  bf16_t* hn_b  = (bf16_t*)alloc((size_t)BB * SP * D * 2);
  bf16_t* Hg_b  = (bf16_t*)alloc((size_t)BB * SP * FF * 2);
  float*  Ewe   = (float*)alloc((size_t)BB * S * D * 4);
  bf16_t* dT_b  = (bf16_t*)alloc((size_t)BB * D * S * 2);
  float*  dsum  = (float*)alloc((size_t)BB * D * 4);
  if (off > ws_size) return;  // insufficient scratch

  const float invV = 1.f / ((float)V + 1e-8f);

  tk_zero<<<2048, 256, 0, stream>>>(fk, BB * SP * D);
  tk_zero<<<8, 256, 0, stream>>>(esum, BB * D);
  tk_zero<<<2, 256, 0, stream>>>(WeCol, D);

  tk_cvt_bf16<<<8192, 256, 0, stream>>>(W_e, We_b, V * D);
  tk_cvt_bf16<<<2048, 256, 0, stream>>>(ff1, w1_b, NL * FF * D);
  tk_cvt_bf16<<<2048, 256, 0, stream>>>(ff2, w2_b, NL * D * FF);
  tk_transpose_bf16<<<dim3(V / 32, D / 32), 256, 0, stream>>>(W_e, WeT_b, V, D);

  tk_embed<<<BB * S, 128, 0, stream>>>(x, W_e, e_f);
  tk_colsum<<<dim3(V / 64, 1), 256, 0, stream>>>(W_e, WeCol, V);
  tk_colsum<<<dim3(S / 64, BB), 256, 0, stream>>>(e_f, esum, S);

  tk_qk_prep<<<dim3(SP, H), 256, 0, stream>>>(W_p, W_q, W_k, Qb, Kb);
  tk_gemm<0><<<dim3(S / 128, 5, H), 128, 0, stream>>>(
      Qb, Kb, scr, nullptr, SP, S, D, D, D, S,
      (long)SP * D, (long)S * D, (long)SP * S);
  tk_softmax<<<H * SP, 256, 0, stream>>>(scr, attn);

  for (int l = 0; l < NL; ++l) {
    if (l == 0) {
      tk_diff_transpose<<<dim3(D / 32, S / 32, BB), 256, 0, stream>>>(
          e_f, nullptr, WeCol, invV, 0, dT_b);
    } else {
      tk_flash_ewe<<<BB * S / 16, 256, 0, stream>>>(fk_b, We_b, WeT_b, Ewe);
      tk_diff_transpose<<<dim3(D / 32, S / 32, BB), 256, 0, stream>>>(
          e_f, Ewe, WeCol, invV, 1, dT_b);
    }
    tk_diffsum<<<BB * D / 8, 256, 0, stream>>>(dT_b, dsum);
    tk_delta_update<<<dim3(D / 16, 33, BB), 32, 0, stream>>>(
        attn, dT_b, W_v + (size_t)l * H * D, A_lr + l * H, dsum, B_lr + l, fk);
    tk_layernorm<<<BB * SP, 256, 0, stream>>>(fk, lng + l * D, hn_b);
    tk_gemm<1><<<dim3(FF / 128, 17, 1), 128, 0, stream>>>(
        hn_b, w1_b + (size_t)l * FF * D, nullptr, Hg_b,
        BB * SP, FF, D, D, D, FF, 0, 0, 0);
    tk_gemm<2><<<dim3(D / 128, 17, 1), 128, 0, stream>>>(
        Hg_b, w2_b + (size_t)l * D * FF, fk, fk_b,
        BB * SP, D, FF, FF, FF, D, 0, 0, 0);
  }

  tk_gemm<0><<<dim3(V / 128, 4, BB), 128, 0, stream>>>(
      fk_b, We_b, out, nullptr, S, V, D, D, D, V,
      (long)SP * D, 0, (long)S * V);

  (void)in_sizes; (void)n_in; (void)out_size;
}
